// GRU_Encoder_19078244729403
// MI455X (gfx1250) — compile-verified, matrix operand order checked by reading, not run on hardware
//
#include <hip/hip_runtime.h>

// ---------------------------------------------------------------------------
// CDNA5 (gfx1250) GRU encoder: bf16 WMMA (f32 accum) for all GEMMs,
// persistent recurrent scan with per-timestep device-wide barrier and
// W_hh staged in LDS via async global->LDS DMA.
// ---------------------------------------------------------------------------

typedef __attribute__((ext_vector_type(16))) __bf16 v16bf;
typedef __attribute__((ext_vector_type(8)))  float  v8f;

struct U32x8 { unsigned int u[8]; };

__device__ __forceinline__ unsigned short f2bf(float f) {
    unsigned int u = __builtin_bit_cast(unsigned int, f);
    unsigned int r = u + 0x7FFFu + ((u >> 16) & 1u);   // round-to-nearest-even
    return (unsigned short)(r >> 16);
}

// A operand: row-major source, lane layout per ISA 16-bit A 16x32 table.
// Memory halves: [off+0..7] then [off+16..23] (off already folded into p).
__device__ __forceinline__ v16bf ldA(const unsigned short* p) {
    uint4 a = *reinterpret_cast<const uint4*>(p);
    uint4 b = *reinterpret_cast<const uint4*>(p + 16);
    U32x8 q;
    q.u[0]=a.x; q.u[1]=a.y; q.u[2]=a.z; q.u[3]=a.w;
    q.u[4]=b.x; q.u[5]=b.y; q.u[6]=b.z; q.u[7]=b.w;
    return __builtin_bit_cast(v16bf, q);
}

// B operand: pre-swizzled packed tile, 16 contiguous halves per lane.
// (Works for both global and LDS pointers; addrspace inferred.)
__device__ __forceinline__ v16bf ldB(const unsigned short* p) {
    uint4 a = *reinterpret_cast<const uint4*>(p);
    uint4 b = *reinterpret_cast<const uint4*>(p + 8);
    U32x8 q;
    q.u[0]=a.x; q.u[1]=a.y; q.u[2]=a.z; q.u[3]=a.w;
    q.u[4]=b.x; q.u[5]=b.y; q.u[6]=b.z; q.u[7]=b.w;
    return __builtin_bit_cast(v16bf, q);
}

__device__ __forceinline__ v8f wmma_bf16(v16bf a, v16bf b, v8f c) {
    return __builtin_amdgcn_wmma_f32_16x16x32_bf16(
        /*neg_a=*/false, a, /*neg_b=*/false, b,
        /*c_mod=*/(short)0, c, /*reuse_a=*/false, /*reuse_b=*/false);
}

__device__ __forceinline__ float sigf(float x)  { return 1.f / (1.f + __expf(-x)); }
__device__ __forceinline__ float tanhf_(float x){ return 1.f - 2.f / (__expf(2.f * x) + 1.f); }

// ---------------------------------------------------------------------------
// Prep: convert x [65536,300] f32 -> bf16 padded to K=320.
// ---------------------------------------------------------------------------
__global__ void conv_x(const float* __restrict__ x, unsigned short* __restrict__ out) {
    size_t tid = (size_t)blockIdx.x * 256 + threadIdx.x;   // 65536*320 threads
    int    k   = (int)(tid % 320);
    size_t m   = tid / 320;
    float  f   = (k < 300) ? x[m * 300 + k] : 0.f;
    out[tid] = f2bf(f);
}

// ---------------------------------------------------------------------------
// Prep: swizzle W [3072, Kreal] f32 row-major into packed bf16 WMMA-B tiles.
// Tile (nt,kt) is 32 lanes x 16 halves; lane l covers col n = nt*16+(l&15),
// k_local = (s<8 ? s : s+8) + (l>=16 ? 8 : 0) for half s.
// ---------------------------------------------------------------------------
__global__ void pack_w(const float* __restrict__ W, unsigned short* __restrict__ out,
                       int Kreal, int KT) {
    int tid  = blockIdx.x * blockDim.x + threadIdx.x;
    int lane = tid & 31;
    int kt   = (tid >> 5) % KT;
    int nt   = (tid >> 5) / KT;
    if (nt >= 192) return;
    int n  = nt * 16 + (lane & 15);
    int ko = (lane >> 4) * 8;
    unsigned int w[8];
    for (int d = 0; d < 8; ++d) {
        int s0 = 2 * d, s1 = 2 * d + 1;
        int k0 = kt * 32 + ((s0 < 8) ? s0 : s0 + 8) + ko;
        int k1 = kt * 32 + ((s1 < 8) ? s1 : s1 + 8) + ko;
        unsigned short h0 = (k0 < Kreal) ? f2bf(W[(size_t)n * Kreal + k0]) : 0;
        unsigned short h1 = (k1 < Kreal) ? f2bf(W[(size_t)n * Kreal + k1]) : 0;
        w[d] = (unsigned int)h0 | ((unsigned int)h1 << 16);
    }
    unsigned int* dst = (unsigned int*)(out + (((size_t)nt * KT + kt) * 32 + lane) * 16);
    for (int d = 0; d < 8; ++d) dst[d] = w[d];
}

// ---------------------------------------------------------------------------
// Input projection: xp[65536,3072] = A[65536,K]*W^T + b_ih + b_hh   (bf16 WMMA)
// One wave computes a 16x64 strip (4 N-tiles), reusing A each k-step.
// ---------------------------------------------------------------------------
__global__ void gemm_xp(const unsigned short* __restrict__ A, int lda, int KT,
                        const unsigned short* __restrict__ Bp,
                        const float* __restrict__ bih, const float* __restrict__ bhh,
                        float* __restrict__ xp) {
    int lane = threadIdx.x & 31;
    int wv   = (blockIdx.x * blockDim.x + threadIdx.x) >> 5;
    int ng   = wv % 48;            // 48 groups of 64 cols
    int mt   = wv / 48;            // 4096 row tiles
    const unsigned short* arow =
        A + (size_t)(mt * 16 + (lane & 15)) * lda + ((lane >> 4) * 8);

    v8f zero = {0.f,0.f,0.f,0.f,0.f,0.f,0.f,0.f};
    v8f acc[4] = {zero, zero, zero, zero};

#pragma unroll 2
    for (int kt = 0; kt < KT; ++kt) {
        v16bf av = ldA(arow + kt * 32);
#pragma unroll
        for (int i = 0; i < 4; ++i) {
            const unsigned short* bp =
                Bp + (((size_t)(ng * 4 + i) * KT + kt) * 32 + lane) * 16;
            acc[i] = wmma_bf16(av, ldB(bp), acc[i]);
        }
    }

    int rbase = mt * 16 + ((lane >> 4) << 3);
#pragma unroll
    for (int i = 0; i < 4; ++i) {
        int col = (ng * 4 + i) * 16 + (lane & 15);
        float bias = bih[col] + bhh[col];
#pragma unroll
        for (int v = 0; v < 8; ++v)
            xp[(size_t)(rbase + v) * 3072 + col] = acc[i][v] + bias;
    }
}

// ---------------------------------------------------------------------------
// Recurrent scan (persistent). 64 blocks x 128 threads; block = jt (column
// tile), wave = mt (batch tile). All 4 waves of a block share the same r/z/n
// B tiles, which are DMA'd into LDS once (global_load_async_to_lds_b128) and
// read via ds_load_b128 every timestep. Double-buffered bf16 h => one
// device-wide barrier per step (per-step counters, 64 arrivals).
// ---------------------------------------------------------------------------
__global__ void gru_scan(const unsigned short* __restrict__ Whh,
                         const float* __restrict__ xp,
                         unsigned short* __restrict__ hbf,   // 2 x 64*1024 bf16
                         float* __restrict__ hf,             // 64*1024 f32
                         unsigned short* __restrict__ out_bf,// layer0 out (bf16)
                         float* __restrict__ out_f,          // layer1 out (f32)
                         float* __restrict__ hlast,          // layer1 h_last
                         int* __restrict__ cnt, int layer1) {
    extern __shared__ unsigned short sB[];   // 3 * 16384 halves = 96 KB
    const int T = 1024, H = 1024;
    int tid   = threadIdx.x;       // 0..127
    int lane  = tid & 31;
    int mt    = tid >> 5;          // wave id = batch tile
    int jt    = blockIdx.x;        // 0..63  = hidden column tile
    int col   = jt * 16 + (lane & 15);
    int rbase = mt * 16 + ((lane >> 4) << 3);
    v8f zero  = {0.f,0.f,0.f,0.f,0.f,0.f,0.f,0.f};

    // ---- One-time async DMA of this block's W_hh slice (r/z/n columns) to LDS.
    {
        unsigned lbase = (unsigned)(uintptr_t)sB;
#pragma unroll
        for (int g = 0; g < 3; ++g) {
            const unsigned short* src = Whh + (size_t)(g * 64 + jt) * 32 * 512;
            for (int i = 0; i < 16; ++i) {             // 32KB / (128 thr * 16B)
                int off = (i * 128 + tid) * 16;        // byte offset in group
                unsigned laddr = lbase + (unsigned)(g * 32768 + off);
                unsigned long long ga =
                    (unsigned long long)(uintptr_t)((const char*)src + off);
                asm volatile("global_load_async_to_lds_b128 %0, %1, off"
                             :: "v"(laddr), "v"(ga) : "memory");
            }
        }
        asm volatile("s_wait_asynccnt 0x0" ::: "memory");
        __syncthreads();
    }

    for (int t = 0; t < T; ++t) {
        if (t > 0) {
            if (tid == 0) {
                volatile int* c = cnt + (t - 1);
                while (*c < 64) __builtin_amdgcn_s_sleep(2);
            }
            __syncthreads();
            __threadfence();   // acquire: see other blocks' h stores
        }

        v8f aR = zero, aZ = zero, aN = zero;
        if (t > 0) {
            const unsigned short* hb   = hbf + (size_t)(t & 1) * 65536;
            const unsigned short* arow =
                hb + (size_t)(mt * 16 + (lane & 15)) * H + ((lane >> 4) * 8);
#pragma unroll 4
            for (int kt = 0; kt < 32; ++kt) {
                v16bf av = ldA(arow + kt * 32);
                v16bf b0 = ldB(sB +             kt * 512 + lane * 16);
                v16bf b1 = ldB(sB + 16384 +     kt * 512 + lane * 16);
                v16bf b2 = ldB(sB + 2 * 16384 + kt * 512 + lane * 16);
                aR = wmma_bf16(av, b0, aR);
                aZ = wmma_bf16(av, b1, aZ);
                aN = wmma_bf16(av, b2, aN);
            }
        }

        unsigned short* hbw = hbf + (size_t)((t + 1) & 1) * 65536;
#pragma unroll
        for (int v = 0; v < 8; ++v) {
            int b = rbase + v;
            const float* xrow = xp + ((size_t)b * T + t) * 3072;
            float r  = sigf (xrow[col]          + aR[v]);
            float zg = sigf (xrow[H + col]      + aZ[v]);
            float n  = tanhf_(xrow[2 * H + col] + r * aN[v]);
            float hp = (t > 0) ? hf[(size_t)b * H + col] : 0.f;
            float hn = (1.f - zg) * n + zg * hp;
            hf [(size_t)b * H + col] = hn;
            hbw[(size_t)b * H + col] = f2bf(hn);
            size_t oi = ((size_t)b * T + t) * H + col;
            if (!layer1) {
                out_bf[oi] = f2bf(hn);
            } else {
                out_f[oi] = hn;
                if (t == T - 1) hlast[(size_t)b * H + col] = hn;
            }
        }

        __threadfence();       // release: publish h before arriving
        __syncthreads();       // all 4 waves done + fenced
        if (tid == 0) atomicAdd(cnt + t, 1);
    }
}

// ---------------------------------------------------------------------------
extern "C" void kernel_launch(void* const* d_in, const int* in_sizes, int n_in,
                              void* d_out, int out_size, void* d_ws, size_t ws_size,
                              hipStream_t stream) {
    (void)in_sizes; (void)n_in; (void)out_size; (void)ws_size;
    const float* x    = (const float*)d_in[0];
    const float* Wih0 = (const float*)d_in[1];
    const float* Whh0 = (const float*)d_in[2];
    const float* bih0 = (const float*)d_in[3];
    const float* bhh0 = (const float*)d_in[4];
    const float* Wih1 = (const float*)d_in[5];
    const float* Whh1 = (const float*)d_in[6];
    const float* bih1 = (const float*)d_in[7];
    const float* bhh1 = (const float*)d_in[8];
    float* out = (float*)d_out;

    char* w = (char*)d_ws;
    size_t o = 0;
    auto alloc = [&](size_t bytes) { size_t r = o; o += (bytes + 255) & ~(size_t)255; return r; };
    int*            cnt0 = (int*)(w + alloc(4096));
    int*            cnt1 = (int*)(w + alloc(4096));
    unsigned short* hbf  = (unsigned short*)(w + alloc((size_t)2 * 65536 * 2));
    float*          hf   = (float*)(w + alloc((size_t)65536 * 4));
    unsigned short* xbf  = (unsigned short*)(w + alloc((size_t)65536 * 320 * 2));
    unsigned short* pIH0 = (unsigned short*)(w + alloc((size_t)3072 * 320  * 2));
    unsigned short* pHH0 = (unsigned short*)(w + alloc((size_t)3072 * 1024 * 2));
    unsigned short* pIH1 = (unsigned short*)(w + alloc((size_t)3072 * 1024 * 2));
    unsigned short* pHH1 = (unsigned short*)(w + alloc((size_t)3072 * 1024 * 2));
    unsigned short* out0 = (unsigned short*)(w + alloc((size_t)65536 * 1024 * 2));
    float*          xp   = (float*)(w + alloc((size_t)65536 * 3072 * 4));

    hipMemsetAsync(cnt0, 0, 4096, stream);
    hipMemsetAsync(cnt1, 0, 4096, stream);

    // Prep: x -> bf16 (padded), weights -> packed bf16 WMMA-B tiles.
    conv_x<<<(65536 * 320) / 256, 256, 0, stream>>>(x, xbf);
    pack_w<<<(192 * 10 * 32) / 256, 256, 0, stream>>>(Wih0, pIH0, 300, 10);
    pack_w<<<(192 * 32 * 32) / 256, 256, 0, stream>>>(Whh0, pHH0, 1024, 32);
    pack_w<<<(192 * 32 * 32) / 256, 256, 0, stream>>>(Wih1, pIH1, 1024, 32);
    pack_w<<<(192 * 32 * 32) / 256, 256, 0, stream>>>(Whh1, pHH1, 1024, 32);

    const int smem = 3 * 32768;   // 96 KB packed W_hh slice per block

    // Layer 0: xp = x*Wih0^T + b_ih0 + b_hh0 ; recurrent scan -> out0 (bf16)
    gemm_xp<<<24576, 256, 0, stream>>>(xbf, 320, 10, pIH0, bih0, bhh0, xp);
    gru_scan<<<64, 128, smem, stream>>>(pHH0, xp, hbf, hf, out0, nullptr, nullptr, cnt0, 0);

    // Layer 1: xp = out0*Wih1^T + b_ih1 + b_hh1 ; scan -> d_out (+ h_last)
    gemm_xp<<<24576, 256, 0, stream>>>(out0, 1024, 32, pIH1, bih1, bhh1, xp);
    gru_scan<<<64, 128, smem, stream>>>(pHH1, xp, hbf, hf, nullptr, out,
                                        out + (size_t)65536 * 1024, cnt1, 1);
}
